// BAG_LSTM_50002009260174
// MI455X (gfx1250) — matrix operand order repaired and checked by reading, not compiled, not run on hardware
//
#include <hip/hip_runtime.h>
#include <hip/hip_bf16.h>

typedef __attribute__((ext_vector_type(16))) _Float16 v16h;
typedef __attribute__((ext_vector_type(8)))  float    v8f;
typedef __attribute__((ext_vector_type(4)))  unsigned int u32x4;
typedef __attribute__((ext_vector_type(4)))  int i32x4;
typedef __attribute__((ext_vector_type(8)))  int i32x8;

#define B_DIM 8192
#define H_DIM 1024
#define K2H   2048
#define N4H   4096

constexpr int BM  = 128;  // rows per workgroup tile
constexpr int KT  = 32;   // K per WMMA step
constexpr int LDA = 40;   // LDS A-tile stride (halfs): 32 data + 8 pad
constexpr int LDB = 40;   // LDS B-tile stride (halfs, K-contiguous rows)

__device__ __forceinline__ float sigmoidf_(float x) { return 1.0f / (1.0f + __expf(-x)); }

__device__ __forceinline__ v8f wmma16(v16h a, v16h b, v8f c) {
  return __builtin_amdgcn_wmma_f32_16x16x32_f16(false, a, false, b, (short)0, c, false, false);
}

// A fragment (16x32 f16, row-major in LDS, K contiguous per row)
__device__ __forceinline__ v16h load_a_frag(const _Float16* As, int lda, int mrow, int lane) {
  int row = mrow + (lane & 15);
  int kb  = (lane >> 4) * 8;
  const _Float16* p = As + row * lda + kb;
  union { v16h h; uint4 q[2]; } u;
  u.q[0] = *(const uint4*)(p);
  u.q[1] = *(const uint4*)(p + 16);
  return u.h;
}

// B fragment (32x16 f16) from K-contiguous (transposed-weight) LDS tile
__device__ __forceinline__ v16h load_b_frag(const _Float16* Bt, int ldb, int ncol, int lane) {
  int n  = ncol + (lane & 15);
  int kb = (lane >> 4) * 8;
  const _Float16* p = Bt + n * ldb + kb;
  union { v16h h; uint4 q[2]; } u;
  u.q[0] = *(const uint4*)(p);
  u.q[1] = *(const uint4*)(p + 16);
  return u.h;
}

// ---------------- Tensor Data Mover: 2D tile global -> LDS (D# per ISA 8.3/8.4) ----
__device__ __forceinline__ void tdm_load_tile_f16(unsigned lds_byte_off,
                                                  const _Float16* gsrc,
                                                  int tile_w_elems, int tile_h,
                                                  int stride_elems) {
  unsigned long long ga = (unsigned long long)(uintptr_t)gsrc;
  u32x4 g0;
  g0.x = 1u;                                                  // count=1 (valid user D#)
  g0.y = lds_byte_off;                                        // lds_addr
  g0.z = (unsigned)(ga & 0xffffffffu);                        // global_addr[31:0]
  g0.w = (unsigned)((ga >> 32) & 0x01ffffffu) | 0x80000000u;  // addr[56:32] | type=2
  i32x8 g1;
  // data_size=1 (2B), pad_enable, pad_interval=3 (16 DWORDs), pad_amount=3 (4 DWORDs)
  g1[0] = (1 << 16) | (1 << 20) | (3 << 22) | (3 << 25);
  g1[1] = (stride_elems & 0xffff) << 16;                       // tensor_dim0 lo
  g1[2] = ((stride_elems >> 16) & 0xffff) | (tile_h << 16);    // tensor_dim0 hi | tensor_dim1 lo
  g1[3] = ((tile_h >> 16) & 0xffff) | (tile_w_elems << 16);    // tensor_dim1 hi | tile_dim0
  g1[4] = (tile_h & 0xffff);                                   // tile_dim1 | tile_dim2=0
  g1[5] = stride_elems;                                        // tensor_dim0_stride lo32
  g1[6] = 0;
  g1[7] = 0;
  i32x4 z4 = {0, 0, 0, 0};
#if defined(__clang_major__) && (__clang_major__ >= 23)
  i32x8 z8 = {0, 0, 0, 0, 0, 0, 0, 0};
  __builtin_amdgcn_tensor_load_to_lds(g0, g1, z4, z4, z8, 0);
#else
  __builtin_amdgcn_tensor_load_to_lds(g0, g1, z4, z4, 0);
#endif
}

// ---------------- conversion kernels ----------------

// transpose [R][C] f32 -> [C][R] f16 (tiled through LDS)
__global__ __launch_bounds__(256) void transpose_cvt(const float* __restrict__ in,
                                                     _Float16* __restrict__ out,
                                                     int R, int C) {
  __shared__ float tile[32][33];
  int bx = blockIdx.x * 32;   // C base
  int by = blockIdx.y * 32;   // R base
  int tx = threadIdx.x & 31;
  int ty = threadIdx.x >> 5;  // 0..7
#pragma unroll
  for (int j = 0; j < 32; j += 8)
    tile[ty + j][tx] = in[(size_t)(by + ty + j) * C + bx + tx];
  __syncthreads();
#pragma unroll
  for (int j = 0; j < 32; j += 8)
    out[(size_t)(bx + ty + j) * R + by + tx] = (_Float16)tile[tx][ty + j];
}

// out[b, 0:H] = x[b], out[b, H:2H] = h0[b]
__global__ __launch_bounds__(256) void concat_cvt(const float* __restrict__ x,
                                                  const float* __restrict__ h,
                                                  _Float16* __restrict__ out) {
  int i   = blockIdx.x * 256 + threadIdx.x;  // over B*2H
  int row = i >> 11;
  int col = i & 2047;
  float v = (col < H_DIM) ? x[row * H_DIM + col] : h[row * H_DIM + (col - H_DIM)];
  out[i] = (_Float16)v;
}

// ---------------- fused LSTM gates GEMM + cell (double-buffered pipeline) ---------
// grid: (B/128, H/32, 2)  block: 256 (8 waves).  WT = W^T [4H][2H]
__global__ __launch_bounds__(256) void lstm_gates_gemm(
    const _Float16* __restrict__ X16a, const _Float16* __restrict__ WTa,
    const float* __restrict__ ba, const float* __restrict__ c0a, const float* __restrict__ mka,
    float* __restrict__ ca, float* __restrict__ oa, _Float16* __restrict__ c16a,
    const _Float16* __restrict__ X16v, const _Float16* __restrict__ WTv,
    const float* __restrict__ bv, const float* __restrict__ c0v, const float* __restrict__ mkv,
    float* __restrict__ cv, float* __restrict__ ov, _Float16* __restrict__ c16v) {
  const bool z = (blockIdx.z != 0);
  const _Float16* X16 = z ? X16v : X16a;
  const _Float16* WT  = z ? WTv : WTa;
  const float* bvec = z ? bv : ba;
  const float* c0   = z ? c0v : c0a;
  const float* mask = z ? mkv : mka;
  float* c_out      = z ? cv : ca;
  float* o_out      = z ? ov : oa;
  _Float16* c16_out = z ? c16v : c16a;

  __shared__ __align__(16) _Float16 As[2][BM * LDA];      // double-buffered A (TDM dest)
  __shared__ __align__(16) _Float16 Bt[2][4 * 32 * LDB];  // double-buffered B (4 gates)

  const int m0 = blockIdx.x * BM;
  const int n0 = blockIdx.y * 32;
  const int tid  = threadIdx.x;
  const int lane = tid & 31;
  const int wave = tid >> 5;
  const int wm = (wave >> 1) * 32;
  const int wn = (wave & 1) * 16;
  const unsigned as_lds[2] = {(unsigned)(uintptr_t)(&As[0][0]),
                              (unsigned)(uintptr_t)(&As[1][0])};

  // this thread's two B chunks (512 chunks total: gate g, column n, 8-half chunk col)
  auto b_src = [&](int c, int k0) -> const uint4* {
    int g = c >> 7, n = (c >> 2) & 31, col = (c & 3) * 8;
    return (const uint4*)(&WT[(size_t)(g * H_DIM + n0 + n) * K2H + k0 + col]);
  };
  auto b_dst = [&](int c, int buf) -> uint4* {
    int g = c >> 7, n = (c >> 2) & 31, col = (c & 3) * 8;
    return (uint4*)(&Bt[buf][(g * 32 + n) * LDB + col]);
  };

  v8f acc[2][4] = {};

  // prologue: stage k0 = 0 into buffer 0
  if (wave == 0) tdm_load_tile_f16(as_lds[0], &X16[(size_t)m0 * K2H], KT, BM, K2H);
  {
    uint4 r0 = *b_src(tid, 0);
    uint4 r1 = *b_src(tid + 256, 0);
    *b_dst(tid, 0) = r0;
    *b_dst(tid + 256, 0) = r1;
  }
  if (wave == 0) __builtin_amdgcn_s_wait_tensorcnt(0);
  __syncthreads();

  int cur = 0;
  for (int k0 = 0; k0 < K2H; k0 += KT) {
    const int nxt = cur ^ 1;
    const bool more = (k0 + KT) < K2H;
    uint4 r0, r1;
    if (more) {
      // async DMA of next A tile + issue next B loads (no wait yet)
      if (wave == 0)
        tdm_load_tile_f16(as_lds[nxt], &X16[(size_t)m0 * K2H + k0 + KT], KT, BM, K2H);
      r0 = *b_src(tid, k0 + KT);
      r1 = *b_src(tid + 256, k0 + KT);
      if (k0 + 2 * KT < K2H)
        __builtin_prefetch(b_src(tid, k0 + 2 * KT), 0, 1);
    }
    // load ALL fragments into distinct registers first (one LDS clause, one wait),
    // then run the 8 WMMAs back-to-back without WAR-induced dscnt stalls.
    const _Float16* Ac = As[cur];
    const _Float16* Bc = Bt[cur];
    v16h a0  = load_a_frag(Ac, LDA, wm, lane);
    v16h a1  = load_a_frag(Ac, LDA, wm + 16, lane);
    v16h bf0 = load_b_frag(&Bc[0 * 32 * LDB], LDB, wn, lane);
    v16h bf1 = load_b_frag(&Bc[1 * 32 * LDB], LDB, wn, lane);
    v16h bf2 = load_b_frag(&Bc[2 * 32 * LDB], LDB, wn, lane);
    v16h bf3 = load_b_frag(&Bc[3 * 32 * LDB], LDB, wn, lane);
    acc[0][0] = wmma16(a0, bf0, acc[0][0]);
    acc[1][0] = wmma16(a1, bf0, acc[1][0]);
    acc[0][1] = wmma16(a0, bf1, acc[0][1]);
    acc[1][1] = wmma16(a1, bf1, acc[1][1]);
    acc[0][2] = wmma16(a0, bf2, acc[0][2]);
    acc[1][2] = wmma16(a1, bf2, acc[1][2]);
    acc[0][3] = wmma16(a0, bf3, acc[0][3]);
    acc[1][3] = wmma16(a1, bf3, acc[1][3]);
    if (more) {
      *b_dst(tid, nxt) = r0;        // loadcnt wait lands here, after WMMAs
      *b_dst(tid + 256, nxt) = r1;
      if (wave == 0) __builtin_amdgcn_s_wait_tensorcnt(0);
    }
    __syncthreads();
    cur = nxt;
  }

  // epilogue: fused LSTM cell
  const int nGlob = n0 + wn + (lane & 15);
  const float bi = bvec[0 * H_DIM + nGlob];
  const float bf = bvec[1 * H_DIM + nGlob];
  const float bg = bvec[2 * H_DIM + nGlob];
  const float bo = bvec[3 * H_DIM + nGlob];
  const int rbase = m0 + wm + ((lane >> 4) ? 8 : 0);
#pragma unroll
  for (int t = 0; t < 2; ++t) {
#pragma unroll
    for (int r = 0; r < 8; ++r) {
      int m = rbase + t * 16 + r;
      float iv = sigmoidf_(acc[t][0][r] + bi);
      float fv = sigmoidf_(acc[t][1][r] + bf);
      float gv = tanhf(acc[t][2][r] + bg);
      float o_ = sigmoidf_(acc[t][3][r] + bo);
      size_t idx = (size_t)m * H_DIM + nGlob;
      float cp = c0[idx];
      float mk = mask[m];
      float cc = fv * cp + iv * gv;
      cc = cc * mk + cp * (1.0f - mk);
      o_ = o_ * mk + (1.0f - mk);
      c_out[idx]   = cc;
      o_out[idx]   = o_;
      c16_out[idx] = (_Float16)cc;
    }
  }
}

// ---------------- fused BAG GEMM, single pipelined K loop -------------------------
// per k-step: acc1 += main x WmbT_top + branch x WmbT_bot ;  acc2 += branch x WbT
// grid: (B/128, H/64, 2)  block: 256.   WmbT [H][2H], WbT [H][H]
__global__ __launch_bounds__(256) void bag_hm_gemm(
    const _Float16* __restrict__ ac16, const _Float16* __restrict__ vc16,
    const _Float16* __restrict__ WmbT, const _Float16* __restrict__ WbT,
    const float* __restrict__ bmb, const float* __restrict__ bb,
    float* __restrict__ hma, float* __restrict__ hmv) {
  const bool z = (blockIdx.z != 0);
  const _Float16* main16   = z ? vc16 : ac16;
  const _Float16* branch16 = z ? ac16 : vc16;
  float* hm_out            = z ? hmv : hma;

  __shared__ __align__(16) _Float16 Am[2][BM * LDA];
  __shared__ __align__(16) _Float16 Ab[2][BM * LDA];
  __shared__ __align__(16) _Float16 B1t[2][64 * LDB];  // WmbT top slice
  __shared__ __align__(16) _Float16 B2t[2][64 * LDB];  // WbT slice
  __shared__ __align__(16) _Float16 B3t[2][64 * LDB];  // WmbT bottom slice

  const int m0 = blockIdx.x * BM;
  const int n0 = blockIdx.y * 64;
  const int tid  = threadIdx.x;
  const int lane = tid & 31;
  const int wave = tid >> 5;
  const int wm = (wave >> 1) * 32;
  const int wn = (wave & 1) * 32;
  const unsigned am_lds[2] = {(unsigned)(uintptr_t)(&Am[0][0]),
                              (unsigned)(uintptr_t)(&Am[1][0])};
  const unsigned ab_lds[2] = {(unsigned)(uintptr_t)(&Ab[0][0]),
                              (unsigned)(uintptr_t)(&Ab[1][0])};

  const int bn  = tid >> 2;          // 0..63  (output column within tile)
  const int bcl = (tid & 3) * 8;     // chunk offset in halfs
  auto src1 = [&](int k0) { return (const uint4*)(&WmbT[(size_t)(n0 + bn) * K2H + k0 + bcl]); };
  auto src3 = [&](int k0) { return (const uint4*)(&WmbT[(size_t)(n0 + bn) * K2H + H_DIM + k0 + bcl]); };
  auto src2 = [&](int k0) { return (const uint4*)(&WbT[(size_t)(n0 + bn) * H_DIM + k0 + bcl]); };

  v8f acc1[2][2] = {};
  v8f acc2[2][2] = {};

  // prologue
  if (wave == 0) {
    tdm_load_tile_f16(am_lds[0], &main16[(size_t)m0 * H_DIM], KT, BM, H_DIM);
    tdm_load_tile_f16(ab_lds[0], &branch16[(size_t)m0 * H_DIM], KT, BM, H_DIM);
  }
  {
    uint4 r1 = *src1(0), r2 = *src2(0), r3 = *src3(0);
    *(uint4*)(&B1t[0][bn * LDB + bcl]) = r1;
    *(uint4*)(&B2t[0][bn * LDB + bcl]) = r2;
    *(uint4*)(&B3t[0][bn * LDB + bcl]) = r3;
  }
  if (wave == 0) __builtin_amdgcn_s_wait_tensorcnt(0);
  __syncthreads();

  int cur = 0;
  for (int k0 = 0; k0 < H_DIM; k0 += KT) {
    const int nxt = cur ^ 1;
    const bool more = (k0 + KT) < H_DIM;
    uint4 r1, r2, r3;
    if (more) {
      if (wave == 0) {
        tdm_load_tile_f16(am_lds[nxt], &main16[(size_t)m0 * H_DIM + k0 + KT], KT, BM, H_DIM);
        tdm_load_tile_f16(ab_lds[nxt], &branch16[(size_t)m0 * H_DIM + k0 + KT], KT, BM, H_DIM);
      }
      r1 = *src1(k0 + KT);
      r2 = *src2(k0 + KT);
      r3 = *src3(k0 + KT);
    }

    // all A fragments + per-u B fragments in distinct registers before WMMA bursts
    const _Float16* Amc = Am[cur];
    const _Float16* Abc = Ab[cur];
    v16h am0 = load_a_frag(Amc, LDA, wm, lane);
    v16h am1 = load_a_frag(Amc, LDA, wm + 16, lane);
    v16h ab0 = load_a_frag(Abc, LDA, wm, lane);
    v16h ab1 = load_a_frag(Abc, LDA, wm + 16, lane);
#pragma unroll
    for (int u = 0; u < 2; ++u) {
      v16h b1 = load_b_frag(B1t[cur], LDB, wn + u * 16, lane);
      v16h b2 = load_b_frag(B2t[cur], LDB, wn + u * 16, lane);
      v16h b3 = load_b_frag(B3t[cur], LDB, wn + u * 16, lane);
      acc1[0][u] = wmma16(am0, b1, acc1[0][u]);
      acc1[1][u] = wmma16(am1, b1, acc1[1][u]);
      acc2[0][u] = wmma16(ab0, b2, acc2[0][u]);
      acc2[1][u] = wmma16(ab1, b2, acc2[1][u]);
      acc1[0][u] = wmma16(ab0, b3, acc1[0][u]);
      acc1[1][u] = wmma16(ab1, b3, acc1[1][u]);
    }
    if (more) {
      *(uint4*)(&B1t[nxt][bn * LDB + bcl]) = r1;
      *(uint4*)(&B2t[nxt][bn * LDB + bcl]) = r2;
      *(uint4*)(&B3t[nxt][bn * LDB + bcl]) = r3;
      if (wave == 0) __builtin_amdgcn_s_wait_tensorcnt(0);
    }
    __syncthreads();
    cur = nxt;
  }

  const int rbase = m0 + wm + ((lane >> 4) ? 8 : 0);
#pragma unroll
  for (int u = 0; u < 2; ++u) {
    int col = n0 + wn + u * 16 + (lane & 15);
    float b1 = bmb[col];
    float b2 = bb[col];
#pragma unroll
    for (int t = 0; t < 2; ++t) {
#pragma unroll
      for (int r = 0; r < 8; ++r) {
        int m = rbase + t * 16 + r;
        float w  = fmaxf(acc1[t][u][r] + b1, 0.0f);
        float hb = acc2[t][u][r] + b2;
        hm_out[(size_t)m * H_DIM + col] = w * hb;
      }
    }
  }
}

// ---------------- per-row finalize ----------------
__device__ __forceinline__ float block_sum(float v, float* buf) {
  int tid = threadIdx.x;
  buf[tid] = v;
  __syncthreads();
  for (int s = 128; s > 0; s >>= 1) {
    if (tid < s) buf[tid] += buf[tid + s];
    __syncthreads();
  }
  float r = buf[0];
  __syncthreads();
  return r;
}

__global__ __launch_bounds__(256) void bag_finalize(
    const float* __restrict__ ca, const float* __restrict__ cv,
    const float* __restrict__ oa, const float* __restrict__ ov,
    const float* __restrict__ hma, const float* __restrict__ hmv,
    const float* __restrict__ is_bag,
    const float* __restrict__ ln_g, const float* __restrict__ ln_b,
    float* __restrict__ out) {
  const int row = blockIdx.x;
  const bool z = (blockIdx.y != 0);
  const float* mainc = z ? cv : ca;
  const float* hm    = z ? hmv : hma;
  const float* ogate = z ? ov : oa;
  float* out_h = out + (size_t)(z ? 2 : 0) * B_DIM * H_DIM + (size_t)row * H_DIM;
  float* out_s = out + (size_t)(z ? 3 : 1) * B_DIM * H_DIM + (size_t)row * H_DIM;

  __shared__ float red[256];
  const float* mrow = mainc + (size_t)row * H_DIM;
  const float* hrow = hm + (size_t)row * H_DIM;

  float mv[4], hv[4];
  float sm2 = 0.0f, sh2 = 0.0f;
#pragma unroll
  for (int j = 0; j < 4; ++j) {
    int col = threadIdx.x + j * 256;
    mv[j] = mrow[col];
    hv[j] = hrow[col];
    sm2 += mv[j] * mv[j];
    sh2 += hv[j] * hv[j];
  }
  float em = sqrtf(block_sum(sm2, red));
  float hn = sqrtf(block_sum(sh2, red));
  if (hn == 0.0f) hn = 1.0f;
  float alpha = fminf(em / (hn + 1e-6f), 1.0f);  // BETA_SHIFT = 1.0

  float pre[4];
  float s1 = 0.0f, s2 = 0.0f;
#pragma unroll
  for (int j = 0; j < 4; ++j) {
    pre[j] = alpha * hv[j] + mv[j];
    s1 += pre[j];
    s2 += pre[j] * pre[j];
  }
  float mu  = block_sum(s1, red) * (1.0f / H_DIM);
  float ex2 = block_sum(s2, red) * (1.0f / H_DIM);
  float var = ex2 - mu * mu;
  float inv = rsqrtf(var + 1e-5f);
  float ib  = is_bag[row];

#pragma unroll
  for (int j = 0; j < 4; ++j) {
    int col = threadIdx.x + j * 256;
    float emb = (pre[j] - mu) * inv * ln_g[col] + ln_b[col];
    float sc  = emb * ib + mv[j] * (1.0f - ib);
    out_s[col] = sc;
    out_h[col] = ogate[(size_t)row * H_DIM + col] * tanhf(sc);
  }
}

// ---------------- host launch ----------------

extern "C" void kernel_launch(void* const* d_in, const int* in_sizes, int n_in,
                              void* d_out, int out_size, void* d_ws, size_t ws_size,
                              hipStream_t stream) {
  (void)in_sizes; (void)n_in; (void)out_size; (void)ws_size;
  const float* a_x  = (const float*)d_in[0];
  const float* a_h0 = (const float*)d_in[1];
  const float* a_c0 = (const float*)d_in[2];
  const float* v_x  = (const float*)d_in[3];
  const float* v_h0 = (const float*)d_in[4];
  const float* v_c0 = (const float*)d_in[5];
  const float* a_mask  = (const float*)d_in[6];
  const float* v_mask  = (const float*)d_in[7];
  const float* is_bag  = (const float*)d_in[8];
  const float* a_W = (const float*)d_in[9];
  const float* a_b = (const float*)d_in[10];
  const float* v_W = (const float*)d_in[11];
  const float* v_b = (const float*)d_in[12];
  const float* W_mb = (const float*)d_in[13];
  const float* b_mb = (const float*)d_in[14];
  const float* W_b  = (const float*)d_in[15];
  const float* b_b  = (const float*)d_in[16];
  const float* ln_g = (const float*)d_in[17];
  const float* ln_b = (const float*)d_in[18];
  float* out = (float*)d_out;

  char* w = (char*)d_ws;
  size_t off = 0;
  auto take = [&](size_t bytes) -> void* {
    void* p = w + off;
    off += (bytes + 255) & ~(size_t)255;
    return p;
  };
  _Float16* aX16  = (_Float16*)take((size_t)B_DIM * K2H * 2);
  _Float16* vX16  = (_Float16*)take((size_t)B_DIM * K2H * 2);
  _Float16* aWT16 = (_Float16*)take((size_t)K2H * N4H * 2);    // [4H][2H]
  _Float16* vWT16 = (_Float16*)take((size_t)K2H * N4H * 2);
  _Float16* WmbT16 = (_Float16*)take((size_t)K2H * H_DIM * 2); // [H][2H]
  _Float16* WbT16  = (_Float16*)take((size_t)H_DIM * H_DIM * 2);
  float* a_c = (float*)take((size_t)B_DIM * H_DIM * 4);
  float* a_o = (float*)take((size_t)B_DIM * H_DIM * 4);
  float* v_c = (float*)take((size_t)B_DIM * H_DIM * 4);
  float* v_o = (float*)take((size_t)B_DIM * H_DIM * 4);
  _Float16* ac16 = (_Float16*)take((size_t)B_DIM * H_DIM * 2);
  _Float16* vc16 = (_Float16*)take((size_t)B_DIM * H_DIM * 2);
  float* hma = (float*)take((size_t)B_DIM * H_DIM * 4);
  float* hmv = (float*)take((size_t)B_DIM * H_DIM * 4);

  // 1) weight transpose+convert (W^T so B fragments are K-contiguous), input packing
  transpose_cvt<<<dim3(N4H / 32, K2H / 32), 256, 0, stream>>>(a_W, aWT16, K2H, N4H);
  transpose_cvt<<<dim3(N4H / 32, K2H / 32), 256, 0, stream>>>(v_W, vWT16, K2H, N4H);
  transpose_cvt<<<dim3(H_DIM / 32, K2H / 32), 256, 0, stream>>>(W_mb, WmbT16, K2H, H_DIM);
  transpose_cvt<<<dim3(H_DIM / 32, H_DIM / 32), 256, 0, stream>>>(W_b, WbT16, H_DIM, H_DIM);
  {
    int nc = B_DIM * K2H;
    concat_cvt<<<(nc + 255) / 256, 256, 0, stream>>>(a_x, a_h0, aX16);
    concat_cvt<<<(nc + 255) / 256, 256, 0, stream>>>(v_x, v_h0, vX16);
  }

  // 2) fused LSTM gate GEMM + cell (both cells via grid.z)
  lstm_gates_gemm<<<dim3(B_DIM / BM, H_DIM / 32, 2), 256, 0, stream>>>(
      aX16, aWT16, a_b, a_c0, a_mask, a_c, a_o, ac16,
      vX16, vWT16, v_b, v_c0, v_mask, v_c, v_o, vc16);

  // 3) fused BAG GEMMs (both directions via grid.z)
  bag_hm_gemm<<<dim3(B_DIM / BM, H_DIM / 64, 2), 256, 0, stream>>>(
      ac16, vc16, WmbT16, WbT16, b_mb, b_b, hma, hmv);

  // 4) per-row norms + LayerNorm + blend + output gate
  bag_finalize<<<dim3(B_DIM, 2), 256, 0, stream>>>(
      a_c, v_c, a_o, v_o, hma, hmv, is_bag, ln_g, ln_b, out);
}
